// PairStr2Pair_3307124817993
// MI455X (gfx1250) — matrix operand order repaired and checked by reading, not compiled, not run on hardware
//
#include <hip/hip_runtime.h>
#include <stdint.h>

// ---------------------------------------------------------------------------
// Types for CDNA5 WMMA (wave32, 16x16x32 f16 -> f32) + TDM descriptors
// ---------------------------------------------------------------------------
typedef _Float16 f16_t;
typedef __attribute__((ext_vector_type(16))) _Float16 v16h;
typedef __attribute__((ext_vector_type(8)))  _Float16 v8h;
typedef __attribute__((ext_vector_type(8)))  float    v8f;
typedef __attribute__((ext_vector_type(4)))  unsigned int u32x4;
typedef __attribute__((ext_vector_type(4)))  int      i32x4;
typedef __attribute__((ext_vector_type(8)))  int      i32x8;

__device__ __forceinline__ v8f wmma16(v16h a, v16h b, v8f c) {
  // emits v_wmma_f32_16x16x32_f16
  return __builtin_amdgcn_wmma_f32_16x16x32_f16(false, a, false, b, (short)0, c,
                                                false, false);
}

// Build a 16-wide fragment from two contiguous 8xf16 (128-bit) loads.
__device__ __forceinline__ v16h ld2(const f16_t* plo, const f16_t* phi) {
  v8h lo = *(const v8h*)plo;
  v8h hi = *(const v8h*)phi;
  return __builtin_shufflevector(lo, hi, 0, 1, 2, 3, 4, 5, 6, 7,
                                 8, 9, 10, 11, 12, 13, 14, 15);
}

__device__ __forceinline__ float fast_sigmoid(float v) {
  return __builtin_amdgcn_rcpf(1.f + __expf(-v));   // v_rcp_f32, no IEEE divide
}

__device__ __forceinline__ float wave_sum(float v) {
  for (int m = 16; m >= 1; m >>= 1) v += __shfl_xor(v, m, 32);
  return v;
}
__device__ __forceinline__ float wave_max(float v) {
  for (int m = 16; m >= 1; m >>= 1) v = fmaxf(v, __shfl_xor(v, m, 32));
  return v;
}

// Crop index map: L=384, crop=256, half=128. Crops cover 128-blocks
// {0,1}, {0,2}, {1,2}.
__device__ __forceinline__ int crop_map(int c, int q) {
  if (c == 0) return q;
  if (c == 1) return (q < 128) ? q : q + 128;
  return q + 128;
}

#define ACT_NONE 0
#define ACT_RELU 1
#define ACT_SIG  2

static constexpr int LL    = 384;
static constexpr int DP    = 128;
static constexpr int CS    = 256;       // crop size
static constexpr int CT    = CS * CS;   // tokens per crop = 65536
static constexpr int NC    = 3;         // number of crops
static constexpr int TPAIR = LL * LL;   // 147456

// ---------------------------------------------------------------------------
// TDM copy: global -> LDS -> global via the Tensor Data Mover, 16KB per wave.
// D# built per CDNA5 ISA: group0 {count=1, lds_addr, global_addr,
// type=2}; group1 {data_size=4B, tensor_dim0=tile_dim0=4096, dims1=1,
// dim0_stride=4096}. Tracked on TENSORcnt.
// ---------------------------------------------------------------------------
__global__ void tdm_copy_kernel(const float* __restrict__ src, float* __restrict__ dst) {
  __shared__ float buf[4096];
  constexpr unsigned TW = 4096;
  unsigned lds = (unsigned)(size_t)(void*)buf;
  unsigned long long ga = (unsigned long long)(size_t)(src + (size_t)blockIdx.x * TW);
  unsigned long long gb = (unsigned long long)(size_t)(dst + (size_t)blockIdx.x * TW);

  i32x8 g1;
  g1[0] = 0x00020000;                       // wg_mask=0, data_size=2 (4B)
  g1[1] = (int)((TW & 0xffffu) << 16);      // tensor_dim0[15:0] in [31:16]
  g1[2] = (int)((TW >> 16) | (1u << 16));   // tensor_dim0[31:16]; tensor_dim1=1
  g1[3] = (int)(TW << 16);                  // tensor_dim1 hi=0; tile_dim0=TW
  g1[4] = 1;                                // tile_dim1=1, tile_dim2=0
  g1[5] = (int)TW;                          // tensor_dim0_stride[31:0]
  g1[6] = 0;                                // stride hi, dim1_stride lo
  g1[7] = 0;

  u32x4 g0l, g0s;
  g0l[0] = 1u;                              // count=1 valid descriptor
  g0l[1] = lds;                             // LDS byte address
  g0l[2] = (unsigned)(ga & 0xffffffffu);
  g0l[3] = (unsigned)((ga >> 32) & 0x01ffffffu) | (2u << 30);  // type=2
  g0s[0] = 1u;
  g0s[1] = lds;
  g0s[2] = (unsigned)(gb & 0xffffffffu);
  g0s[3] = (unsigned)((gb >> 32) & 0x01ffffffu) | (2u << 30);

  i32x4 z4 = {0, 0, 0, 0};
#if defined(__clang_major__) && __clang_major__ >= 23
  i32x8 z8 = {0, 0, 0, 0, 0, 0, 0, 0};
  __builtin_amdgcn_tensor_load_to_lds(g0l, g1, z4, z4, z8, 0);
  __builtin_amdgcn_s_wait_tensorcnt(0);
  __builtin_amdgcn_tensor_store_from_lds(g0s, g1, z4, z4, z8, 0);
#else
  __builtin_amdgcn_tensor_load_to_lds(g0l, g1, z4, z4, 0);
  __builtin_amdgcn_s_wait_tensorcnt(0);
  __builtin_amdgcn_tensor_store_from_lds(g0s, g1, z4, z4, 0);
#endif
  __builtin_amdgcn_s_wait_tensorcnt(0);
}

// ---------------------------------------------------------------------------
// Elementwise / setup kernels
// ---------------------------------------------------------------------------
// Transpose + convert one weight matrix: W f32 [K][N] -> Wt f16 [N][K]
__global__ void wt_tr_kernel(const float* __restrict__ W, f16_t* __restrict__ Wt,
                             int K, int N) {
  int i = blockIdx.x * 256 + threadIdx.x;
  if (i >= K * N) return;
  int k = i / N, n = i % N;
  Wt[(size_t)n * K + k] = (f16_t)W[i];
}

// out = g * x  (g f16 gate already activated, x f32); flat layout
__global__ void ew_mul_kernel(const f16_t* __restrict__ g, const float* __restrict__ x,
                              void* __restrict__ out, int out_f16, int n) {
  int i = blockIdx.x * 256 + threadIdx.x;
  if (i >= n) return;
  float v = (float)g[i] * x[i];
  if (out_f16) ((f16_t*)out)[i] = (f16_t)v;
  else         ((float*)out)[i] = v;
}

// Gated multiply + transpose into per-d-slice row-major [d][row][col] layout
// consumed by tri_einsum with contiguous-k fragments.
// swap=0 (outgoing, tok=(i,k)): T[d][a][b];  swap=1 (incoming, tok=(k,i)): T[d][b][a]
__global__ void ew_mul_tr_kernel(const f16_t* __restrict__ g, const float* __restrict__ x,
                                 f16_t* __restrict__ out, int swap) {
  int i = blockIdx.x * 256 + threadIdx.x;
  if (i >= CT * 32) return;
  int tok = i >> 5, d = i & 31;
  int a = tok >> 8, b = tok & 255;
  int r = swap ? b : a;
  int c = swap ? a : b;
  out[((size_t)d * 256 + r) * 256 + c] = (f16_t)((float)g[i] * x[i]);
}

// LN(state) -> left/right projections (384 tokens, 32 -> 16)
__global__ void state_proj_kernel(const float* __restrict__ state,
                                  const float* __restrict__ g, const float* __restrict__ b,
                                  const float* __restrict__ wl, const float* __restrict__ bl,
                                  const float* __restrict__ wr, const float* __restrict__ br,
                                  float* __restrict__ left, float* __restrict__ right) {
  int wave = blockIdx.x * 4 + (threadIdx.x >> 5);
  int lane = threadIdx.x & 31;
  if (wave >= LL) return;
  float x = state[wave * 32 + lane];
  float s = wave_sum(x), s2 = wave_sum(x * x);
  float mean = s * (1.f / 32.f);
  float var  = s2 * (1.f / 32.f) - mean * mean;
  float rs   = rsqrtf(var + 1e-5f);
  float xn   = (x - mean) * rs * g[lane] + b[lane];
  for (int j = 0; j < 16; j++) {
    float pl = wave_sum(xn * wl[lane * 16 + j]);
    float pr = wave_sum(xn * wr[lane * 16 + j]);
    if (lane == 0) {
      left [wave * 16 + j] = pl + bl[j];
      right[wave * 16 + j] = pr + br[j];
    }
  }
}

// Generic layernorm over last dim D (32 or 128), f32 in -> f16 out
__global__ void ln_kernel(const float* __restrict__ x, const float* __restrict__ g,
                          const float* __restrict__ b, f16_t* __restrict__ y,
                          int T, int D) {
  int wave = blockIdx.x * 4 + (threadIdx.x >> 5);
  int lane = threadIdx.x & 31;
  if (wave >= T) return;
  const float* xr = x + (size_t)wave * D;
  int cnt = D >> 5;
  float v[4], s = 0.f, s2 = 0.f;
  for (int r = 0; r < cnt; r++) { float t = xr[lane + 32 * r]; v[r] = t; s += t; s2 += t * t; }
  s = wave_sum(s); s2 = wave_sum(s2);
  float mean = s / (float)D;
  float var  = s2 / (float)D - mean * mean;
  float rs   = rsqrtf(var + 1e-5f);
  f16_t* yr = y + (size_t)wave * D;
  for (int r = 0; r < cnt; r++) {
    int f = lane + 32 * r;
    yr[f] = (f16_t)((v[r] - mean) * rs * g[f] + b[f]);
  }
}

// Gather crop token (c,i,j) from pair (optional transpose) + layernorm -> f16
__global__ void ln_gather_kernel(const float* __restrict__ pairw,
                                 const float* __restrict__ g, const float* __restrict__ b,
                                 f16_t* __restrict__ out, int transpose) {
  int wave = blockIdx.x * 4 + (threadIdx.x >> 5);   // c*65536 + i*256 + j
  int lane = threadIdx.x & 31;
  int c = wave >> 16;
  int ij = wave & 65535;
  int i = ij >> 8, j = ij & 255;
  int l = crop_map(c, transpose ? j : i);
  int m = crop_map(c, transpose ? i : j);
  const float* xr = pairw + ((size_t)l * LL + m) * DP;
  float v[4], s = 0.f, s2 = 0.f;
  for (int r = 0; r < 4; r++) { float t = xr[lane + 32 * r]; v[r] = t; s += t; s2 += t * t; }
  s = wave_sum(s); s2 = wave_sum(s2);
  float mean = s * (1.f / 128.f);
  float var  = s2 * (1.f / 128.f) - mean * mean;
  float rs   = rsqrtf(var + 1e-5f);
  f16_t* yr = out + (size_t)wave * DP;
  for (int r = 0; r < 4; r++) {
    int f = lane + 32 * r;
    yr[f] = (f16_t)((v[r] - mean) * rs * g[f] + b[f]);
  }
}

// Attention bias: gather rbf (f16), LN(norm_bias), dot with b_w (128x4)
__global__ void bias_kernel(const f16_t* __restrict__ rbf,
                            const float* __restrict__ g, const float* __restrict__ b,
                            const float* __restrict__ bw, float* __restrict__ out,
                            int c, int transpose) {
  int wave = blockIdx.x * 4 + (threadIdx.x >> 5);   // i*256 + j
  int lane = threadIdx.x & 31;
  int i = wave >> 8, j = wave & 255;
  int l = crop_map(c, transpose ? j : i);
  int m = crop_map(c, transpose ? i : j);
  const f16_t* xr = rbf + ((size_t)l * LL + m) * DP;
  float v[4], s = 0.f, s2 = 0.f;
  for (int r = 0; r < 4; r++) { float t = (float)xr[lane + 32 * r]; v[r] = t; s += t; s2 += t * t; }
  s = wave_sum(s); s2 = wave_sum(s2);
  float mean = s * (1.f / 128.f);
  float var  = s2 * (1.f / 128.f) - mean * mean;
  float rs   = rsqrtf(var + 1e-5f);
  float xn[4];
  for (int r = 0; r < 4; r++) {
    int f = lane + 32 * r;
    xn[r] = (v[r] - mean) * rs * g[f] + b[f];
  }
  for (int h = 0; h < 4; h++) {
    float p = 0.f;
    for (int r = 0; r < 4; r++) p += xn[r] * bw[(lane + 32 * r) * 4 + h];
    p = wave_sum(p);
    if (lane == 0) out[(size_t)wave * 4 + h] = p;
  }
}

// Scatter-add delta/cnt back into pair state (atomic f32; crops overlap)
__global__ void scatter_add_kernel(float* __restrict__ pairw, const float* __restrict__ delta,
                                   int c, int transpose) {
  int tok = blockIdx.x;
  int ch  = threadIdx.x;
  int i = tok >> 8, j = tok & 255;
  int l = crop_map(c, transpose ? j : i);
  int m = crop_map(c, transpose ? i : j);
  float cnt = ((l >> 7) == (m >> 7)) ? 2.f : 1.f;
  float v = delta[(size_t)tok * DP + ch] / cnt;
  __hip_atomic_fetch_add(&pairw[((size_t)l * LL + m) * DP + ch], v,
                         __ATOMIC_RELAXED, __HIP_MEMORY_SCOPE_AGENT);
}

// ---------------------------------------------------------------------------
// Generic token GEMM: Y = act(scale*(X@Wt^T) + bias) [+ resid]
// X f16 [T][K], Wt f16 [N][K] (pre-transposed weights). Both fragments are
// per-lane k-contiguous -> 2x global_load_b128 each. No LDS, no syncs.
// TRANS: write Y as f16 [N][T] (for the attention V matrix).
// ---------------------------------------------------------------------------
template <int ACT, bool OF16, bool TRANS, bool RESID>
__global__ void gemm_kernel(const f16_t* __restrict__ X, const f16_t* __restrict__ Wt,
                            const float* __restrict__ bias, const float* __restrict__ resid,
                            void* __restrict__ Y, int T, int K, int N, float scale) {
  const int m0   = blockIdx.x * 16;
  const int w    = threadIdx.x >> 5;
  const int lane = threadIdx.x & 31;
  const int n0   = blockIdx.y * 64 + w * 16;
  const int col  = lane & 15;
  const int arow = lane & 15;
  const int kb   = (lane >> 4) * 8;
  const int bkb  = (lane >> 4) * 16;
  if (n0 >= N) return;
  const f16_t* ap = X  + (size_t)(m0 + arow) * K;
  const f16_t* bp = Wt + (size_t)(n0 + col) * K;
  v8f acc = {0.f, 0.f, 0.f, 0.f, 0.f, 0.f, 0.f, 0.f};
  for (int kk = 0; kk < K; kk += 32) {
    v16h a = ld2(ap + kk + kb, ap + kk + 16 + kb);
    v16h b = ld2(bp + kk + kb, bp + kk + 16 + kb);
    acc = wmma16(a, b, acc);
  }
  const int cg = n0 + col;
  const float bb = bias ? bias[cg] : 0.f;
#pragma unroll
  for (int r = 0; r < 8; r++) {
    int row = m0 + (lane >> 4) * 8 + r;
    float v = acc[r] * scale + bb;
    if (ACT == ACT_RELU)     v = fmaxf(v, 0.f);
    else if (ACT == ACT_SIG) v = fast_sigmoid(v);
    size_t o = TRANS ? ((size_t)cg * T + row) : ((size_t)row * N + cg);
    if (RESID) v += resid[o];
    if (OF16) ((f16_t*)Y)[o] = (f16_t)v;
    else      ((float*)Y)[o] = v;
  }
}

// ---------------------------------------------------------------------------
// Gated RBF embedding: rbf = sigmoid(outer(left,right)@Wg + bg) * (feat@We + be)
// Weights pre-transposed f16 [128][K].
// ---------------------------------------------------------------------------
__global__ void gate_rbf_kernel(const f16_t* __restrict__ WgT, const float* __restrict__ bg,
                                const f16_t* __restrict__ WeT, const float* __restrict__ be,
                                const float* __restrict__ left, const float* __restrict__ right,
                                const float* __restrict__ rbf_feat, f16_t* __restrict__ out) {
  __shared__ f16_t feat[16][264];   // 528B rows -> 16B aligned
  __shared__ f16_t rbff[16][72];    // 144B rows -> 16B aligned
  const int p0   = blockIdx.x * 16;
  const int w    = threadIdx.x >> 5;
  const int lane = threadIdx.x & 31;
  const int n0   = blockIdx.y * 64 + w * 16;
  const int col  = lane & 15;
  const int arow = lane & 15;
  const int kb   = (lane >> 4) * 8;
  const int bkb  = (lane >> 4) * 16;

  for (int e = threadIdx.x; e < 16 * 256; e += 128) {
    int p = e >> 8, f = e & 255;
    int pos = p0 + p;
    int l = pos / LL, m = pos % LL;
    feat[p][f] = (f16_t)(left[l * 16 + (f >> 4)] * right[m * 16 + (f & 15)]);
  }
  for (int e = threadIdx.x; e < 16 * 64; e += 128) {
    int p = e >> 6, f = e & 63;
    rbff[p][f] = (f16_t)rbf_feat[(size_t)(p0 + p) * 64 + f];
  }
  __syncthreads();

  const f16_t* bg_p = WgT + (size_t)(n0 + col) * 256;
  v8f acc1 = {0.f, 0.f, 0.f, 0.f, 0.f, 0.f, 0.f, 0.f};
  for (int kk = 0; kk < 256; kk += 32) {
    v16h a = ld2(&feat[arow][kk + kb], &feat[arow][kk + 16 + kb]);
    v16h b = ld2(bg_p + kk + kb, bg_p + kk + 16 + kb);
    acc1 = wmma16(a, b, acc1);
  }
  const f16_t* be_p = WeT + (size_t)(n0 + col) * 64;
  v8f acc2 = {0.f, 0.f, 0.f, 0.f, 0.f, 0.f, 0.f, 0.f};
  for (int kk = 0; kk < 64; kk += 32) {
    v16h a = ld2(&rbff[arow][kk + kb], &rbff[arow][kk + 16 + kb]);
    v16h b = ld2(be_p + kk + kb, be_p + kk + 16 + kb);
    acc2 = wmma16(a, b, acc2);
  }
  const int cg = n0 + col;
#pragma unroll
  for (int r = 0; r < 8; r++) {
    int pos = p0 + (lane >> 4) * 8 + r;
    float gate = fast_sigmoid(acc1[r] + bg[cg]);
    out[(size_t)pos * DP + cg] = (f16_t)(gate * (acc2[r] + be[cg]));
  }
}

// ---------------------------------------------------------------------------
// Triangle einsum on pre-transposed slices: out[i,j,d] = (1/256) sum_k
//   TL[d][i][k] * TR[d][j][k].  All fragments 128-bit contiguous loads.
// ---------------------------------------------------------------------------
__global__ void tri_einsum_kernel(const f16_t* __restrict__ TL, const f16_t* __restrict__ TR,
                                  float* __restrict__ out) {
  const int d    = blockIdx.z;
  const int i0   = blockIdx.x * 16;
  const int w    = threadIdx.x >> 5;
  const int lane = threadIdx.x & 31;
  const int j0   = (blockIdx.y * 4 + w) * 16;
  const int arow = lane & 15;
  const int bcol = lane & 15;
  const int kb   = (lane >> 4) * 8;
  const int bkb  = (lane >> 4) * 16;
  const f16_t* ap = TL + ((size_t)d * 256 + i0 + arow) * 256;
  const f16_t* bp = TR + ((size_t)d * 256 + j0 + bcol) * 256;
  v8f acc = {0.f, 0.f, 0.f, 0.f, 0.f, 0.f, 0.f, 0.f};
  for (int kk = 0; kk < 256; kk += 32) {
    v16h a = ld2(ap + kk + kb, ap + kk + 16 + kb);
    v16h b = ld2(bp + kk + kb, bp + kk + 16 + kb);
    acc = wmma16(a, b, acc);
  }
#pragma unroll
  for (int r = 0; r < 8; r++) {
    int row = i0 + (lane >> 4) * 8 + r;
    out[((size_t)row * 256 + (j0 + bcol)) * 32 + d] = acc[r] * (1.f / 256.f);
  }
}

// ---------------------------------------------------------------------------
// Attention scores: attn[h,i,j] = sum_{n,k} q[n,i,h,k] * kmat[n,j,h,k]
// ---------------------------------------------------------------------------
__global__ void attn_score_kernel(const f16_t* __restrict__ q, const f16_t* __restrict__ kmat,
                                  float* __restrict__ attn) {
  const int h    = blockIdx.z;
  const int i0   = blockIdx.x * 16;
  const int w    = threadIdx.x >> 5;
  const int lane = threadIdx.x & 31;
  const int j0   = (blockIdx.y * 4 + w) * 16;
  const int arow = lane & 15;
  const int bcol = lane & 15;
  const int kb   = (lane >> 4) * 8;
  const int bkb  = (lane >> 4) * 16;
  const size_t aoff = (size_t)(i0 + arow) * DP + h * 32;
  const size_t boff = (size_t)(j0 + bcol) * DP + h * 32;
  v8f acc = {0.f, 0.f, 0.f, 0.f, 0.f, 0.f, 0.f, 0.f};
  for (int n = 0; n < 256; n++) {
    const f16_t* ap = q + (size_t)n * 256 * DP + aoff;
    const f16_t* bp = kmat + (size_t)n * 256 * DP + boff;
    v16h a = ld2(ap + kb, ap + 16 + kb);
    v16h b = ld2(bp + kb, bp + 16 + kb);
    acc = wmma16(a, b, acc);
  }
#pragma unroll
  for (int r = 0; r < 8; r++) {
    int row = i0 + (lane >> 4) * 8 + r;
    attn[(size_t)h * CT + row * 256 + (j0 + bcol)] = acc[r];
  }
}

// softmax over j with fused bias; f32 in, f16 out. wave per (h,i)
__global__ void softmax_kernel(const float* __restrict__ attn, const float* __restrict__ bias,
                               f16_t* __restrict__ out) {
  int wave = blockIdx.x * 4 + (threadIdx.x >> 5);  // h*256 + i
  int lane = threadIdx.x & 31;
  int h = wave >> 8, i = wave & 255;
  const float* ar = attn + (size_t)h * CT + i * 256;
  float v[8], mx = -1e30f;
  for (int t = 0; t < 8; t++) {
    int j = lane + 32 * t;
    v[t] = ar[j] + bias[(size_t)(i * 256 + j) * 4 + h];
    mx = fmaxf(mx, v[t]);
  }
  mx = wave_max(mx);
  float s = 0.f;
  for (int t = 0; t < 8; t++) { v[t] = __expf(v[t] - mx); s += v[t]; }
  s = wave_sum(s);
  float inv = __builtin_amdgcn_rcpf(s);
  f16_t* o = out + (size_t)h * CT + i * 256;
  for (int t = 0; t < 8; t++) o[lane + 32 * t] = (f16_t)(v[t] * inv);
}

// out[n,i,h*32+d] = sum_j attnS[h,i,j] * vt[h*32+d][n*256+j]
// vt is the GEMM-transposed V: f16 [128][65536]
__global__ void attn_out_kernel(const f16_t* __restrict__ attnS, const f16_t* __restrict__ vt,
                                float* __restrict__ out) {
  const int h    = blockIdx.z >> 1;
  const int d0   = (blockIdx.z & 1) * 16;
  const int w    = threadIdx.x >> 5;
  const int lane = threadIdx.x & 31;
  const int n    = blockIdx.y * 4 + w;
  const int i0   = blockIdx.x * 16;
  const int arow = lane & 15;
  const int bcol = lane & 15;
  const int kb   = (lane >> 4) * 8;
  const int bkb  = (lane >> 4) * 16;
  const f16_t* ap = attnS + (size_t)h * CT + (i0 + arow) * 256;
  const f16_t* bp = vt + (size_t)(h * 32 + d0 + bcol) * CT + (size_t)n * 256;
  v8f acc = {0.f, 0.f, 0.f, 0.f, 0.f, 0.f, 0.f, 0.f};
  for (int kk = 0; kk < 256; kk += 32) {
    v16h a = ld2(ap + kk + kb, ap + kk + 16 + kb);
    v16h b = ld2(bp + kk + kb, bp + kk + 16 + kb);
    acc = wmma16(a, b, acc);
  }
#pragma unroll
  for (int r = 0; r < 8; r++) {
    int row = i0 + (lane >> 4) * 8 + r;
    out[((size_t)n * 256 + row) * DP + h * 32 + d0 + bcol] = acc[r];
  }
}

// ---------------------------------------------------------------------------
// Workspace layout (bytes). Per-op working set ~100-200MB -> fits 192MB L2.
// ---------------------------------------------------------------------------
static constexpr size_t O_PAIR  = 0;                                  // f32 pair state
static constexpr size_t SZ_PAIR = (size_t)TPAIR * DP * 4;
static constexpr size_t O_RBF   = O_PAIR + SZ_PAIR;                   // f16 gated rbf
static constexpr size_t SZ_RBF  = (size_t)TPAIR * DP * 2;
static constexpr size_t O_X     = O_RBF + SZ_RBF;                     // f16 gathered LN input
static constexpr size_t SZ_X    = (size_t)NC * CT * DP * 2;
static constexpr size_t SZ_F16  = (size_t)CT * DP * 2;
static constexpr size_t O_F16A  = O_X + SZ_X;
static constexpr size_t O_F16B  = O_F16A + SZ_F16;
static constexpr size_t O_F16C  = O_F16B + SZ_F16;
static constexpr size_t O_F16D  = O_F16C + SZ_F16;
static constexpr size_t SZ_F32  = (size_t)CT * DP * 4;
static constexpr size_t O_F32A  = O_F16D + SZ_F16;
static constexpr size_t O_F32B  = O_F32A + SZ_F32;
static constexpr size_t O_SATTN  = O_F32B + SZ_F32;                   // f32 [4][65536]
static constexpr size_t O_SATTNS = O_SATTN + (size_t)4 * CT * 4;      // f16 [4][65536]
static constexpr size_t O_SBIAS  = O_SATTNS + (size_t)4 * CT * 2;     // f32 [65536][4]
static constexpr size_t O_SLEFT  = O_SBIAS + (size_t)CT * 4 * 4;
static constexpr size_t O_SRIGHT = O_SLEFT + (size_t)LL * 16 * 4;
static constexpr size_t O_WT     = O_SRIGHT + (size_t)LL * 16 * 4;    // f16 transposed weights

extern "C" void kernel_launch(void* const* d_in, const int* in_sizes, int n_in,
                              void* d_out, int out_size, void* d_ws, size_t ws_size,
                              hipStream_t stream) {
  (void)in_sizes; (void)out_size; (void)ws_size;
  const float* in_pair  = (const float*)d_in[0];
  const float* in_rbf   = (const float*)d_in[1];
  const float* in_state = (const float*)d_in[2];
  const float* P[128];
  for (int i = 0; i < n_in && i < 128; i++) P[i] = (const float*)d_in[i];

  // Flattened params (recursive dict-insertion order) -- see round 1 notes.
  const int TRI0 = 13, TRI1 = 29, RATT = 45, CATT = 57, FFB = 69;

  char* ws = (char*)d_ws;
  float* A_pair  = (float*)(ws + O_PAIR);
  f16_t* A_rbf   = (f16_t*)(ws + O_RBF);
  f16_t* A_X     = (f16_t*)(ws + O_X);
  f16_t* A_f16a  = (f16_t*)(ws + O_F16A);
  f16_t* A_f16b  = (f16_t*)(ws + O_F16B);
  f16_t* A_f16c  = (f16_t*)(ws + O_F16C);
  f16_t* A_f16d  = (f16_t*)(ws + O_F16D);
  float* A_f32a  = (float*)(ws + O_F32A);
  float* A_f32b  = (float*)(ws + O_F32B);
  float* S_attn  = (float*)(ws + O_SATTN);
  f16_t* S_attnS = (f16_t*)(ws + O_SATTNS);
  float* S_bias  = (float*)(ws + O_SBIAS);
  float* S_left  = (float*)(ws + O_SLEFT);
  float* S_right = (float*)(ws + O_SRIGHT);
  f16_t* FF_X    = (f16_t*)(ws + O_F32A);  // FFN reuses dead arenas
  f16_t* FF_H    = (f16_t*)(ws + O_X);

  // ---- pre-transpose all GEMM weights to f16 [N][K] (tiny, L2-resident) ----
  f16_t* WT[128] = {nullptr};
  {
    f16_t* wbase = (f16_t*)(ws + O_WT);
    size_t woff = 0;
    auto prep = [&](int idx, int K, int N) {
      WT[idx] = wbase + woff;
      wt_tr_kernel<<<(K * N + 255) / 256, 256, 0, stream>>>(P[idx], WT[idx], K, N);
      woff += (size_t)K * N;
    };
    prep(9, 256, 128);    // to_gate
    prep(11, 64, 128);    // emb_rbf
    for (int base : {TRI0, TRI1}) {
      prep(base + 4, 128, 32);   prep(base + 6, 128, 32);
      prep(base + 8, 128, 32);   prep(base + 10, 128, 32);
      prep(base + 12, 128, 128); prep(base + 14, 32, 128);
    }
    for (int base : {RATT, CATT}) {
      prep(base + 4, 128, 128);  prep(base + 5, 128, 128);
      prep(base + 6, 128, 128);  prep(base + 8, 128, 128);
      prep(base + 10, 128, 128);
    }
    prep(FFB + 2, 128, 256);  prep(FFB + 4, 256, 128);
  }

  // Specialized GEMM launchers (weights via WT)
  auto gdim = [](int T, int N) { return dim3(T / 16, (N + 63) / 64); };
  auto gemm_f16 = [&](const f16_t* X, int widx, const float* bias, f16_t* Y,
                      int T, int K, int N, float sc) {
    gemm_kernel<ACT_NONE, true, false, false>
        <<<gdim(T, N), 128, 0, stream>>>(X, WT[widx], bias, nullptr, Y, T, K, N, sc);
  };
  auto gemm_vtr = [&](const f16_t* X, int widx, f16_t* Y, int T, int K, int N) {
    gemm_kernel<ACT_NONE, true, true, false>
        <<<gdim(T, N), 128, 0, stream>>>(X, WT[widx], nullptr, nullptr, Y, T, K, N, 1.f);
  };
  auto gemm_sig = [&](const f16_t* X, int widx, const float* bias, f16_t* Y,
                      int T, int K, int N) {
    gemm_kernel<ACT_SIG, true, false, false>
        <<<gdim(T, N), 128, 0, stream>>>(X, WT[widx], bias, nullptr, Y, T, K, N, 1.f);
  };
  auto gemm_f32 = [&](const f16_t* X, int widx, const float* bias, float* Y,
                      int T, int K, int N) {
    gemm_kernel<ACT_NONE, false, false, false>
        <<<gdim(T, N), 128, 0, stream>>>(X, WT[widx], bias, nullptr, Y, T, K, N, 1.f);
  };
  auto gemm_relu = [&](const f16_t* X, int widx, const float* bias, f16_t* Y,
                       int T, int K, int N) {
    gemm_kernel<ACT_RELU, true, false, false>
        <<<gdim(T, N), 128, 0, stream>>>(X, WT[widx], bias, nullptr, Y, T, K, N, 1.f);
  };
  auto gemm_res = [&](const f16_t* X, int widx, const float* bias, const float* resid,
                      float* Y, int T, int K, int N) {
    gemm_kernel<ACT_NONE, false, false, true>
        <<<gdim(T, N), 128, 0, stream>>>(X, WT[widx], bias, resid, Y, T, K, N, 1.f);
  };
  auto ew = [&](const f16_t* g, const float* x, void* o, int of16, int n) {
    ew_mul_kernel<<<(n + 255) / 256, 256, 0, stream>>>(g, x, o, of16, n);
  };

  // ---- init: pair state via TDM DMA, state projections, gated rbf ----
  tdm_copy_kernel<<<SZ_PAIR / 16384, 32, 0, stream>>>(in_pair, A_pair);
  state_proj_kernel<<<96, 128, 0, stream>>>(in_state, P[3], P[4], P[5], P[6], P[7], P[8],
                                            S_left, S_right);
  gate_rbf_kernel<<<dim3(TPAIR / 16, 2), 128, 0, stream>>>(WT[9], P[10], WT[11], P[12],
                                                           S_left, S_right, in_rbf, A_rbf);

  // ---- triangle multiplications (outgoing then incoming) ----
  auto tri_pass = [&](int base, int outgoing) {
    int swap = outgoing ? 0 : 1;
    ln_gather_kernel<<<NC * CT / 4, 128, 0, stream>>>(A_pair, P[base + 0], P[base + 1], A_X, 0);
    for (int c = 0; c < NC; c++) {
      const f16_t* Xc = A_X + (size_t)c * CT * DP;
      gemm_sig(Xc, base + 8,  P[base + 9],  A_f16b, CT, 128, 32);               // left_gate
      gemm_f32(Xc, base + 4,  P[base + 5],  A_f32a, CT, 128, 32);               // left_proj
      ew_mul_tr_kernel<<<(CT * 32 + 255) / 256, 256, 0, stream>>>(A_f16b, A_f32a, A_f16a, swap);
      gemm_sig(Xc, base + 10, P[base + 11], A_f16b, CT, 128, 32);               // right_gate
      gemm_f32(Xc, base + 6,  P[base + 7],  A_f32a, CT, 128, 32);               // right_proj
      ew_mul_tr_kernel<<<(CT * 32 + 255) / 256, 256, 0, stream>>>(A_f16b, A_f32a, A_f16c, swap);
      tri_einsum_kernel<<<dim3(16, 4, 32), 128, 0, stream>>>(A_f16a, A_f16c, A_f32a);
      ln_kernel<<<CT / 4, 128, 0, stream>>>(A_f32a, P[base + 2], P[base + 3], A_f16b, CT, 32);
      gemm_f32(A_f16b, base + 14, P[base + 15], A_f32b, CT, 32, 128);           // out_proj
      gemm_sig(Xc,     base + 12, P[base + 13], A_f16d, CT, 128, 128);          // gate
      ew(A_f16d, A_f32b, A_f32a, 0, CT * 128);                                  // delta f32
      scatter_add_kernel<<<CT, 128, 0, stream>>>(A_pair, A_f32a, c, 0);
    }
  };
  tri_pass(TRI0, 1);
  tri_pass(TRI1, 0);

  // ---- axial attention (row: transpose=1, col: transpose=0) ----
  auto attn_pass = [&](int base, int transpose) {
    ln_gather_kernel<<<NC * CT / 4, 128, 0, stream>>>(A_pair, P[base + 0], P[base + 1], A_X,
                                                      transpose);
    for (int c = 0; c < NC; c++) {
      const f16_t* Xc = A_X + (size_t)c * CT * DP;
      gemm_f16(Xc, base + 4, nullptr, A_f16a, CT, 128, 128, 0.1767766953f);     // q
      gemm_f16(Xc, base + 5, nullptr, A_f16b, CT, 128, 128, 1.f / 256.f);       // k
      gemm_vtr(Xc, base + 6, A_f16c, CT, 128, 128);                             // v (transposed)
      gemm_sig(Xc, base + 8, P[base + 9], A_f16d, CT, 128, 128);                // gate
      bias_kernel<<<CT / 4, 128, 0, stream>>>(A_rbf, P[base + 2], P[base + 3], P[base + 7],
                                              S_bias, c, transpose);
      attn_score_kernel<<<dim3(16, 4, 4), 128, 0, stream>>>(A_f16a, A_f16b, S_attn);
      softmax_kernel<<<256, 128, 0, stream>>>(S_attn, S_bias, S_attnS);
      attn_out_kernel<<<dim3(16, 64, 8), 128, 0, stream>>>(S_attnS, A_f16c, A_f32a);
      ew(A_f16d, A_f32a, A_f16a, 1, CT * 128);                                  // gate*out
      gemm_f32(A_f16a, base + 10, P[base + 11], A_f32b, CT, 128, 128);          // out proj
      scatter_add_kernel<<<CT, 128, 0, stream>>>(A_pair, A_f32b, c, transpose);
    }
  };
  attn_pass(RATT, 1);
  attn_pass(CATT, 0);

  // ---- FFN at full resolution, residual fused into final GEMM -> d_out ----
  ln_kernel<<<TPAIR / 4, 128, 0, stream>>>(A_pair, P[FFB + 0], P[FFB + 1], FF_X, TPAIR, 128);
  gemm_relu(FF_X, FFB + 2, P[FFB + 3], FF_H, TPAIR, 128, 256);
  gemm_res(FF_H, FFB + 4, P[FFB + 5], A_pair, (float*)d_out, TPAIR, 256, 128);
}
